// GNN_Encoder_77146202571445
// MI455X (gfx1250) — matrix operand order repaired and checked by reading, not compiled, not run on hardware
//
#include <hip/hip_runtime.h>
#include <hip/hip_bf16.h>

// GNN encoder: 2x (GCNConv -> BN -> ReLU) on MI455X (gfx1250, wave32, WMMA).
// GEMMs: v_wmma_f32_16x16x32_bf16 with pre-swizzled bf16 operands, fp32 acc,
// double-buffered LDS staging via async global->LDS copies (ASYNCcnt).
// Aggregation via f32 atomics (51MB agg buffer is L2-resident, 192MB L2).

#define NN 50000
#define EE 800000
#define DD 256
#define KT 8      // K tiles of 32  (256/32)
#define NT 16     // N tiles of 16  (256/16)
#define MT 3125   // M tiles of 16  (50000/16)
#define BN_EPS 1e-5f

typedef __attribute__((ext_vector_type(16))) __bf16 v16bf;
typedef __attribute__((ext_vector_type(8)))  float  v8f;

__device__ __forceinline__ unsigned pack_bf16(float a, float b) {
  unsigned ua = __float_as_uint(a), ub = __float_as_uint(b);
  ua = (ua + 0x7FFFu + ((ua >> 16) & 1u)) >> 16;   // round-to-nearest-even
  ub = (ub + 0x7FFFu + ((ub >> 16) & 1u)) >> 16;
  return (ub << 16) | (ua & 0xFFFFu);
}

// Async 16B global -> LDS copy (gfx1250 GLOBAL_LOAD_ASYNC_TO_LDS_B128).
// Inline asm keeps it portable across ROCm/upstream builtin arity skew.
__device__ __forceinline__ void async_cp16(void* lds, const void* g) {
  unsigned loff = (unsigned)(unsigned long long)lds;   // addr[31:0] = LDS offset
  asm volatile("global_load_async_to_lds_b128 %0, %1, off"
               :: "v"(loff), "v"(g) : "memory");
}
__device__ __forceinline__ void wait_async0() {
  asm volatile("s_wait_asynccnt 0x0" ::: "memory");
}

// ---------------- degree / normalization ----------------
__global__ void k_init_deg(float* deg) {
  int i = blockIdx.x * 256 + threadIdx.x;
  if (i < NN) deg[i] = 1.0f;                        // self-loop
}
__global__ void k_count_deg(const int* __restrict__ col, float* deg) {
  int e = blockIdx.x * 256 + threadIdx.x;
  if (e < EE) atomicAdd(&deg[col[e]], 1.0f);
}
__global__ void k_rsqrt_deg(float* deg) {
  int i = blockIdx.x * 256 + threadIdx.x;
  if (i < NN) deg[i] = rsqrtf(deg[i]);
}

// ---------------- bf16 swizzled packing -----------------
// A-fragment layout (16-bit A 16x32, ISA 7.12.2):
//   m = lane&15, khalf = lane>>4, VGPR v holds K = (v>>2)*16 + khalf*8 + (v&3)*2 (+1)
// Optionally fuses BN scale/shift + ReLU (apply==1).
__global__ void __launch_bounds__(256)
k_pack_A(const float* __restrict__ src, const float* __restrict__ scsh,
         int apply, unsigned* __restrict__ Asw) {
  int idx = blockIdx.x * 256 + threadIdx.x;         // MT*KT*32 = 800000 exact
  int lane = idx & 31, tmp = idx >> 5;
  int kt = tmp & 7, mt = tmp >> 3;
  if (mt >= MT) return;
  int m = (mt << 4) + (lane & 15);
  int khalf = lane >> 4;
  const float* srow = src + ((size_t)m << 8) + (kt << 5);
  unsigned o[8];
#pragma unroll
  for (int v = 0; v < 8; ++v) {
    int k = ((v >> 2) << 4) + (khalf << 3) + ((v & 3) << 1);
    float f0 = srow[k], f1 = srow[k + 1];
    if (apply) {
      int ch = (kt << 5) + k;
      f0 = fmaxf(fmaf(f0, scsh[ch],     scsh[DD + ch]),     0.0f);
      f1 = fmaxf(fmaf(f1, scsh[ch + 1], scsh[DD + ch + 1]), 0.0f);
    }
    o[v] = pack_bf16(f0, f1);
  }
  uint4* d4 = (uint4*)(Asw + ((size_t)((mt * KT + kt) * 32 + lane)) * 8);
  d4[0] = make_uint4(o[0], o[1], o[2], o[3]);
  d4[1] = make_uint4(o[4], o[5], o[6], o[7]);
}

// B-fragment layout (16-bit B 32x16): n = lane&15, VGPR v holds
//   K = (v>>2)*16 + (lane>>4)*8 + (v&3)*2 (+1)
__global__ void k_pack_B(const float* __restrict__ W, unsigned* __restrict__ Bsw) {
  int idx = blockIdx.x * 256 + threadIdx.x;         // KT*NT*32 = 4096
  if (idx >= KT * NT * 32) return;
  int lane = idx & 31, tmp = idx >> 5;
  int nt = tmp & 15, kt = tmp >> 4;
  int n = (nt << 4) + (lane & 15);
  int khalf = lane >> 4;
  unsigned o[8];
#pragma unroll
  for (int v = 0; v < 8; ++v) {
    int ch = (kt << 5) + ((v >> 2) << 4) + (khalf << 3) + ((v & 3) << 1);
    float f0 = W[(size_t)ch * DD + n];
    float f1 = W[(size_t)(ch + 1) * DD + n];
    o[v] = pack_bf16(f0, f1);
  }
  uint4* d4 = (uint4*)(Bsw + ((size_t)((kt * NT + nt) * 32 + lane)) * 8);
  d4[0] = make_uint4(o[0], o[1], o[2], o[3]);
  d4[1] = make_uint4(o[4], o[5], o[6], o[7]);
}

// ---------------- WMMA GEMM: H = A(bf16) x B(bf16), f32 acc ----------------
// Block = 8 waves = 2 m-tiles x 4 n-quads (32x256 output patch).
// Per K-step an 18KB slab (B: 16 tiles, A: 2 tiles) is staged into LDS with
// async global->LDS b128 copies, double-buffered so copy overlaps WMMA math.
__global__ void __launch_bounds__(256)
gemm_wmma_bf16(const unsigned* __restrict__ Aswu, const unsigned* __restrict__ Bswu,
               float* __restrict__ H) {
  __shared__ unsigned Bsh[2][NT * 32 * 8];   // 2 x 16KB
  __shared__ unsigned Ash[2][2 * 32 * 8];    // 2 x  2KB
  const int tid  = threadIdx.x;
  const int wave = tid >> 5, lane = tid & 31;
  const int wmt  = wave >> 2;                       // 0..1: m-tile within block
  const int nq   = wave & 3;                        // n-quad (4 tiles of 16)
  const int mt   = blockIdx.x * 2 + wmt;
  const int mt0  = blockIdx.x * 2;

  // stage slab for K-step kt into buffer buf (all 256 threads participate)
  auto stage = [&](int buf, int kt) {
    const unsigned* gB = Bswu + (size_t)kt * (NT * 32 * 8);
#pragma unroll
    for (int i = 0; i < 4; ++i) {                   // B: 1024 x 16B
      int c = tid + i * 256;
      async_cp16(&Bsh[buf][c * 4], gB + (size_t)c * 4);
    }
    if (tid < 128) {                                // A: 128 x 16B (2 tiles)
      int t = tid >> 6, c = tid & 63;
      int am = mt0 + t; if (am > MT - 1) am = MT - 1;   // tail-block clamp
      const unsigned* gA = Aswu + ((size_t)(am * KT + kt) * 32) * 8 + c * 4;
      async_cp16(&Ash[buf][(t * 64 + c) * 4], gA);
    }
  };

  stage(0, 0);
  wait_async0();
  __syncthreads();

  v8f acc0 = {}, acc1 = {}, acc2 = {}, acc3 = {};
#pragma unroll
  for (int kt = 0; kt < KT; ++kt) {
    int cur = kt & 1;
    if (kt < KT - 1) stage(cur ^ 1, kt + 1);        // async copy overlaps math
    const v16bf* Ac = (const v16bf*)&Ash[cur][0];
    const v16bf* Bc = (const v16bf*)&Bsh[cur][0];
    v16bf a  = Ac[wmt * 32 + lane];
    v16bf b0 = Bc[(nq * 4 + 0) * 32 + lane];
    v16bf b1 = Bc[(nq * 4 + 1) * 32 + lane];
    v16bf b2 = Bc[(nq * 4 + 2) * 32 + lane];
    v16bf b3 = Bc[(nq * 4 + 3) * 32 + lane];
    acc0 = __builtin_amdgcn_wmma_f32_16x16x32_bf16(false, a, false, b0, (short)0, acc0, false, false);
    acc1 = __builtin_amdgcn_wmma_f32_16x16x32_bf16(false, a, false, b1, (short)0, acc1, false, false);
    acc2 = __builtin_amdgcn_wmma_f32_16x16x32_bf16(false, a, false, b2, (short)0, acc2, false, false);
    acc3 = __builtin_amdgcn_wmma_f32_16x16x32_bf16(false, a, false, b3, (short)0, acc3, false, false);
    wait_async0();                                  // next slab landed
    __syncthreads();                                // all waves done with cur
  }

  if (mt < MT) {
    // C/D layout: VGPR r -> row (lane>>4)*8 + r, col lane&15
    int m0 = (mt << 4) + ((lane >> 4) << 3);
    int c0 = (nq << 6) + (lane & 15);
    float* hb = H + (size_t)m0 * DD + c0;
#pragma unroll
    for (int r = 0; r < 8; ++r) {
      hb[(size_t)r * DD + 0]  = acc0[r];
      hb[(size_t)r * DD + 16] = acc1[r];
      hb[(size_t)r * DD + 32] = acc2[r];
      hb[(size_t)r * DD + 48] = acc3[r];
    }
  }
}

// ---------------- aggregation ----------------
// agg = h * dinv[i]^2 (self loop) + bias
__global__ void k_init_agg(const float* __restrict__ h, const float* __restrict__ dinv,
                           const float* __restrict__ bias, float* __restrict__ agg) {
  size_t idx = (size_t)blockIdx.x * 256 + threadIdx.x;    // NN*DD threads
  int i = (int)(idx >> 8), d = (int)(idx & 255);
  float di = dinv[i];
  agg[idx] = h[idx] * di * di + bias[d];
}

// agg[col] += h[row] * dinv[row]*dinv[col]  — 64 lanes/edge, float4 gather
__global__ void __launch_bounds__(256)
k_scatter(const int* __restrict__ row, const int* __restrict__ col,
          const float* __restrict__ dinv, const float* __restrict__ h,
          float* __restrict__ agg) {
  int e = (blockIdx.x << 2) + (threadIdx.x >> 6);
  if (e >= EE) return;
  int t = threadIdx.x & 63;
  int r = row[e], c = col[e];
  float w = dinv[r] * dinv[c];
  const float4 hv = *((const float4*)(h + ((size_t)r << 8)) + t);
  float* dst = agg + ((size_t)c << 8) + (t << 2);
  atomicAdd(dst + 0, hv.x * w);
  atomicAdd(dst + 1, hv.y * w);
  atomicAdd(dst + 2, hv.z * w);
  atomicAdd(dst + 3, hv.w * w);
}

// ---------------- batch norm ----------------
__global__ void k_zero_stats(float* stats) {
  int i = threadIdx.x;
  if (i < 2 * DD) stats[i] = 0.0f;
}
__global__ void __launch_bounds__(256)
k_bn_stats(const float* __restrict__ agg, float* __restrict__ stats) {
  int d = threadIdx.x;
  int r0 = blockIdx.x * 250;                              // 200 blocks * 250 rows
  float s = 0.0f, q = 0.0f;
  for (int r = r0; r < r0 + 250; ++r) {
    float v = agg[((size_t)r << 8) + d];
    s += v; q += v * v;
  }
  atomicAdd(&stats[d], s);
  atomicAdd(&stats[DD + d], q);
}
__global__ void k_bn_finalize(float* stats, const float* __restrict__ gamma,
                              const float* __restrict__ beta) {
  int d = threadIdx.x;
  float m = stats[d] * (1.0f / NN);
  float v = stats[DD + d] * (1.0f / NN) - m * m;
  float sc = rsqrtf(v + BN_EPS) * gamma[d];
  stats[d] = sc;                                          // scale
  stats[DD + d] = beta[d] - m * sc;                       // shift
}
__global__ void k_bn_apply_out(const float* __restrict__ agg,
                               const float* __restrict__ stats,
                               float* __restrict__ out) {
  size_t idx = (size_t)blockIdx.x * 256 + threadIdx.x;
  int d = (int)(idx & 255);
  out[idx] = fmaxf(fmaf(agg[idx], stats[d], stats[DD + d]), 0.0f);
}

// ---------------- launch ----------------
extern "C" void kernel_launch(void* const* d_in, const int* in_sizes, int n_in,
                              void* d_out, int out_size, void* d_ws, size_t ws_size,
                              hipStream_t stream) {
  (void)in_sizes; (void)n_in; (void)out_size; (void)ws_size;
  const float* x   = (const float*)d_in[0];
  const int*   ei  = (const int*)d_in[1];       // int32 (JAX demotes int64)
  const float* W1  = (const float*)d_in[2];
  const float* b1  = (const float*)d_in[3];
  const float* g1  = (const float*)d_in[4];
  const float* be1 = (const float*)d_in[5];
  const float* W2  = (const float*)d_in[6];
  const float* b2  = (const float*)d_in[7];
  const float* g2  = (const float*)d_in[8];
  const float* be2 = (const float*)d_in[9];
  const int* row = ei;
  const int* col = ei + EE;

  // workspace carve-out (256B aligned): ~128.5 MB total
  char* base = (char*)d_ws;
  size_t off = 0;
  auto take = [&](size_t bytes) -> void* {
    void* p = base + off;
    off += (bytes + 255) & ~(size_t)255;
    return p;
  };
  float*    dinv  = (float*)take((size_t)NN * 4);
  unsigned* Asw   = (unsigned*)take((size_t)NN * DD * 2);
  unsigned* Bsw1  = (unsigned*)take((size_t)DD * DD * 2);
  unsigned* Bsw2  = (unsigned*)take((size_t)DD * DD * 2);
  float*    h     = (float*)take((size_t)NN * DD * 4);
  float*    agg   = (float*)take((size_t)NN * DD * 4);
  float*    stats = (float*)take((size_t)2 * DD * 4);

  dim3 blk(256);
  const int packA_blocks = (MT * KT * 32) / 256;          // 3125
  const int gemm_blocks  = (MT + 1) / 2;                  // 1563 (2 m-tiles/block)
  const int nd_blocks    = NN;                            // NN*DD/256

  // degree / dinv
  k_init_deg<<<(NN + 255) / 256, blk, 0, stream>>>(dinv);
  k_count_deg<<<(EE + 255) / 256, blk, 0, stream>>>(col, dinv);
  k_rsqrt_deg<<<(NN + 255) / 256, blk, 0, stream>>>(dinv);

  // pack weights (once)
  k_pack_B<<<16, blk, 0, stream>>>(W1, Bsw1);
  k_pack_B<<<16, blk, 0, stream>>>(W2, Bsw2);

  // ---- layer 1 ----
  k_pack_A<<<packA_blocks, blk, 0, stream>>>(x, nullptr, 0, Asw);
  gemm_wmma_bf16<<<gemm_blocks, blk, 0, stream>>>(Asw, Bsw1, h);
  k_init_agg<<<nd_blocks, blk, 0, stream>>>(h, dinv, b1, agg);
  k_scatter<<<EE / 4, blk, 0, stream>>>(row, col, dinv, h, agg);
  k_zero_stats<<<1, 512, 0, stream>>>(stats);
  k_bn_stats<<<200, blk, 0, stream>>>(agg, stats);
  k_bn_finalize<<<1, blk, 0, stream>>>(stats, g1, be1);
  k_pack_A<<<packA_blocks, blk, 0, stream>>>(agg, stats, 1, Asw);  // BN+ReLU fused

  // ---- layer 2 ----
  gemm_wmma_bf16<<<gemm_blocks, blk, 0, stream>>>(Asw, Bsw2, h);
  k_init_agg<<<nd_blocks, blk, 0, stream>>>(h, dinv, b2, agg);
  k_scatter<<<EE / 4, blk, 0, stream>>>(row, col, dinv, h, agg);
  k_zero_stats<<<1, 512, 0, stream>>>(stats);
  k_bn_stats<<<200, blk, 0, stream>>>(agg, stats);
  k_bn_finalize<<<1, blk, 0, stream>>>(stats, g2, be2);
  k_bn_apply_out<<<nd_blocks, blk, 0, stream>>>(agg, stats, (float*)d_out);
}
